// PathwayFusionSNN_48275432407475
// MI455X (gfx1250) — matrix operand order repaired and checked by reading, not compiled, not run on hardware
//
#include <hip/hip_runtime.h>
#include <hip/hip_fp16.h>

// ---------------------------------------------------------------------------
// PathwayFusionSNN forward for MI455X (gfx1250, wave32, WMMA).
// GEMMs: f16 WMMA, f32 accumulate. Data path: async global->LDS staging
// (ASYNCcnt) with double buffering, fragments read from LDS (ds_load_b128).
// ---------------------------------------------------------------------------

typedef __attribute__((ext_vector_type(16))) _Float16 v16h;
typedef __attribute__((ext_vector_type(8)))  _Float16 v8h;
typedef __attribute__((ext_vector_type(8)))  float    v8f;
typedef __attribute__((ext_vector_type(4)))  int      v4i;

#define SNN_BETA 0.9f
#define SNN_THR  1.0f

// LDS tile geometry: 128 rows x 32 halves, padded to 40 halves/row so that
// 16 consecutive rows hit distinct banks (row stride = 20 DWORDs, gcd-free).
#define TILE_ROWS 128
#define TILE_K    32
#define ROW_PAD   40

// ----------------------- CDNA5 async-copy primitives -----------------------

__device__ __forceinline__ void async_copy_b128(const _Float16* gsrc,
                                                _Float16* ldst) {
#if __has_builtin(__builtin_amdgcn_global_load_async_to_lds_b128)
  __builtin_amdgcn_global_load_async_to_lds_b128(
      (__attribute__((address_space(1))) v4i*)gsrc,
      (__attribute__((address_space(3))) v4i*)ldst, 0, 0);
#else
  unsigned lds_off =
      (unsigned)(uintptr_t)(__attribute__((address_space(3))) void*)ldst;
  asm volatile("global_load_async_to_lds_b128 %0, %1, off" ::"v"(lds_off),
               "v"(gsrc)
               : "memory");
#endif
}

template <int N>
__device__ __forceinline__ void wait_asynccnt() {
#if __has_builtin(__builtin_amdgcn_s_wait_asynccnt)
  __builtin_amdgcn_s_wait_asynccnt((unsigned short)N);
#else
  asm volatile("s_wait_asynccnt %0" ::"i"(N));
#endif
}

// ------------------------------- elementwise -------------------------------

__global__ __launch_bounds__(256) void cvt_f32_to_f16_k(
    const float* __restrict__ in, _Float16* __restrict__ out,
    float scale, long n) {
  long i = (long)blockIdx.x * blockDim.x + threadIdx.x;
  long stride = (long)gridDim.x * blockDim.x;
  for (; i < n; i += stride) out[i] = (_Float16)(in[i] * scale);
}

// LIF membrane update for the fusion layer: produces f16 spike matrix.
__global__ __launch_bounds__(256) void lif_fusion_step_k(
    const float* __restrict__ fcur, float* __restrict__ fmem,
    _Float16* __restrict__ fspk, long n) {
  long i = (long)blockIdx.x * blockDim.x + threadIdx.x;
  if (i >= n) return;
  float old   = fmem[i];
  float reset = (old > SNN_THR) ? SNN_THR : 0.0f;
  float nm    = SNN_BETA * old + fcur[i] - reset;
  fmem[i] = nm;
  fspk[i] = (_Float16)((nm > SNN_THR) ? 1.0f : 0.0f);
}

// ------------------------------- WMMA GEMM ---------------------------------
// C[M,N] = A[M,K] (f16 row-major) * W[N,K]^T (f16 row-major) + bias
// Block: 128 threads = 4 waves (2x2), each wave owns 64x64 = 4x4 WMMA tiles.
// grid.x = M/128, grid.y = N/128.  M,N multiples of 128; K multiple of 32.
// A/B tiles staged global->LDS with async copies, double-buffered on ASYNCcnt.
//
// EPI=0: out f16 = relu(v)         (pathway GEMMs -> fused buffer slice)
// EPI=1: out f32 = v               (fusion current / final output)
// EPI=2: fused LIF integration epilogue (imem update + spike -> pooled sum)

template <int EPI>
__global__ __launch_bounds__(128) void wmma_gemm_k(
    const _Float16* __restrict__ A, int lda,
    const _Float16* __restrict__ W, int ldw,
    const float* __restrict__ bias, int K,
    _Float16* __restrict__ outH, float* __restrict__ outF, int ldc, int colOff,
    float* __restrict__ imem, float* __restrict__ pooled, int ldmem) {
  __shared__ _Float16 sA[2][TILE_ROWS * ROW_PAD];
  __shared__ _Float16 sB[2][TILE_ROWS * ROW_PAD];

  const int tid   = threadIdx.x;
  const int lane  = tid & 31;
  const int wave  = tid >> 5;
  const int wrow  = wave >> 1;           // 0..1
  const int wcol  = wave & 1;            // 0..1
  const int rowB  = blockIdx.x * 128;    // block row origin in A/C
  const int colB  = blockIdx.y * 128;    // block col origin in W/C
  const int ml    = lane & 15;           // M (A) / N (B,C,D) index within tile
  const int kh    = (lane >> 4) & 1;     // half-wave selector

  // Issue one stage of async copies: A tile + B tile (8 x b128 per thread).
  auto stage_issue = [&](int buf, int kk) {
#pragma unroll
    for (int it = 0; it < 4; ++it) {
      const int ch = tid + it * 128;       // 512 chunks of 16B for A
      const int r = ch >> 2, c = ch & 3;   // 4 chunks per 32-half row
      async_copy_b128(A + (size_t)(rowB + r) * lda + kk + c * 8,
                      &sA[buf][r * ROW_PAD + c * 8]);
    }
#pragma unroll
    for (int it = 0; it < 4; ++it) {
      const int ch = tid + it * 128;
      const int r = ch >> 2, c = ch & 3;
      async_copy_b128(W + (size_t)(colB + r) * ldw + kk + c * 8,
                      &sB[buf][r * ROW_PAD + c * 8]);
    }
  };

  v8f acc[4][4];
#pragma unroll
  for (int mi = 0; mi < 4; ++mi)
#pragma unroll
    for (int ni = 0; ni < 4; ++ni) {
      v8f z = {};
      acc[mi][ni] = z;
    }

  const int nstages = K / TILE_K;
  stage_issue(0, 0);

  for (int st = 0; st < nstages; ++st) {
    const int cur = st & 1;
    if (st + 1 < nstages) {
      stage_issue(1 - cur, (st + 1) * TILE_K);
      wait_asynccnt<8>();   // newest 8 = next stage; current stage retired
    } else {
      wait_asynccnt<0>();
    }
    __syncthreads();        // publish LDS writes of all waves

    v16h afr[4];
    v16h bfr[4];
    // A fragment (16x32 f16): lane ml = row, two 16B runs at K = kh*8 (+16).
#pragma unroll
    for (int mi = 0; mi < 4; ++mi) {
      const _Float16* p =
          &sA[cur][(wrow * 64 + mi * 16 + ml) * ROW_PAD + kh * 8];
      v8h lo = *(const v8h*)(p);
      v8h hi = *(const v8h*)(p + 16);
      afr[mi] = __builtin_shufflevector(lo, hi, 0, 1, 2, 3, 4, 5, 6, 7, 8, 9,
                                        10, 11, 12, 13, 14, 15);
    }
    // B fragment (32x16 f16): lane ml = column, two 16B runs at
    // K = kh*8 (+16).
#pragma unroll
    for (int ni = 0; ni < 4; ++ni) {
      const _Float16* p =
          &sB[cur][(wcol * 64 + ni * 16 + ml) * ROW_PAD + kh * 8];
      v8h lo = *(const v8h*)(p);
      v8h hi = *(const v8h*)(p + 16);
      bfr[ni] = __builtin_shufflevector(lo, hi, 0, 1, 2, 3, 4, 5, 6, 7, 8, 9,
                                        10, 11, 12, 13, 14, 15);
    }
#pragma unroll
    for (int mi = 0; mi < 4; ++mi)
#pragma unroll
      for (int ni = 0; ni < 4; ++ni)
        acc[mi][ni] = __builtin_amdgcn_wmma_f32_16x16x32_f16(
            false, afr[mi], false, bfr[ni], (short)0, acc[mi][ni], false,
            false);

    __syncthreads();        // all reads of `cur` done before it is refilled
  }

  // Epilogue. C/D layout: lane -> n = ml, vgpr r -> m = r + 8*kh.
#pragma unroll
  for (int mi = 0; mi < 4; ++mi) {
#pragma unroll
    for (int ni = 0; ni < 4; ++ni) {
      const int n  = colB + wcol * 64 + ni * 16 + ml;
      const float bv = bias[n];
#pragma unroll
      for (int r = 0; r < 8; ++r) {
        const int m = rowB + wrow * 64 + mi * 16 + r + kh * 8;
        float v = acc[mi][ni][r] + bv;
        if (EPI == 0) {
          outH[(size_t)m * ldc + colOff + n] =
              (_Float16)(v > 0.0f ? v : 0.0f);
        } else if (EPI == 1) {
          outF[(size_t)m * ldc + n] = v;
        } else {  // EPI == 2 : LIF integration layer fused epilogue
          const size_t idx = (size_t)m * ldmem + n;
          float old   = imem[idx];
          float reset = (old > SNN_THR) ? SNN_THR : 0.0f;
          float nm    = SNN_BETA * old + v - reset;
          imem[idx]   = nm;
          float spk   = (nm > SNN_THR) ? 1.0f : 0.0f;
          pooled[idx] += spk;
        }
      }
    }
  }
}

// ------------------------------- launcher ----------------------------------

extern "C" void kernel_launch(void* const* d_in, const int* in_sizes, int n_in,
                              void* d_out, int out_size, void* d_ws,
                              size_t ws_size, hipStream_t stream) {
  (void)in_sizes; (void)n_in; (void)out_size; (void)ws_size;
  constexpr size_t B = 4096, D = 256, BH = 1024, H = 2048, O = 128;
  constexpr int NUM_STEPS = 16;

  const float* distance = (const float*)d_in[0];
  const float* azimuth  = (const float*)d_in[1];
  const float* elev     = (const float*)d_in[2];
  // d_in[3] = spike_count: unused by the reference output.
  const float* Wd = (const float*)d_in[4];  const float* bd = (const float*)d_in[5];
  const float* Wa = (const float*)d_in[6];  const float* ba = (const float*)d_in[7];
  const float* We = (const float*)d_in[8];  const float* be = (const float*)d_in[9];
  const float* Wf = (const float*)d_in[10]; const float* bf = (const float*)d_in[11];
  const float* Wi = (const float*)d_in[12]; const float* bi = (const float*)d_in[13];
  const float* Wo = (const float*)d_in[14]; const float* bo = (const float*)d_in[15];
  float* out = (float*)d_out;

  // Workspace carving (256-byte aligned regions).
  char* ws = (char*)d_ws;
  size_t off = 0;
  auto carve = [&](size_t bytes) {
    size_t o = off;
    off += (bytes + 255) & ~(size_t)255;
    return o;
  };
  float*    fcur     = (float*)(ws + carve(B * H * 4));
  float*    fmem     = (float*)(ws + carve(B * H * 4));
  float*    imem     = (float*)(ws + carve(B * H * 4));
  float*    pooled   = (float*)(ws + carve(B * H * 4));
  _Float16* fspk     = (_Float16*)(ws + carve(B * H * 2));
  _Float16* fused    = (_Float16*)(ws + carve(B * 3 * BH * 2));
  _Float16* pooled16 = (_Float16*)(ws + carve(B * H * 2));
  _Float16* X16      = (_Float16*)(ws + carve(3 * B * D * 2));
  _Float16* Wd16     = (_Float16*)(ws + carve(BH * D * 2));
  _Float16* Wa16     = (_Float16*)(ws + carve(BH * D * 2));
  _Float16* We16     = (_Float16*)(ws + carve(BH * D * 2));
  _Float16* Wf16     = (_Float16*)(ws + carve(H * 3 * BH * 2));
  _Float16* Wi16     = (_Float16*)(ws + carve(H * H * 2));
  _Float16* Wo16     = (_Float16*)(ws + carve(O * H * 2));

  // Zero state (graph-capture safe).
  (void)hipMemsetAsync(fmem, 0, B * H * 4, stream);
  (void)hipMemsetAsync(imem, 0, B * H * 4, stream);
  (void)hipMemsetAsync(pooled, 0, B * H * 4, stream);

  auto cvt = [&](const float* src, _Float16* dst, float scale, long n) {
    int blocks = (int)((n + 255) / 256);
    cvt_f32_to_f16_k<<<blocks, 256, 0, stream>>>(src, dst, scale, n);
  };
  cvt(distance, X16 + 0 * B * D, 1.0f, (long)(B * D));
  cvt(azimuth,  X16 + 1 * B * D, 1.0f, (long)(B * D));
  cvt(elev,     X16 + 2 * B * D, 1.0f, (long)(B * D));
  cvt(Wd, Wd16, 1.0f, (long)(BH * D));
  cvt(Wa, Wa16, 1.0f, (long)(BH * D));
  cvt(We, We16, 1.0f, (long)(BH * D));
  cvt(Wf, Wf16, 1.0f, (long)(H * 3 * BH));
  cvt(Wi, Wi16, 1.0f, (long)(H * H));
  cvt(Wo, Wo16, 1.0f, (long)(O * H));

  // Stage 1: three pathway GEMMs with ReLU -> fused [B, 3*BH] (f16).
  {
    dim3 grid((unsigned)(B / 128), (unsigned)(BH / 128));
    const _Float16* Wp[3] = {Wd16, Wa16, We16};
    const float*    bp[3] = {bd, ba, be};
    for (int p = 0; p < 3; ++p) {
      wmma_gemm_k<0><<<grid, 128, 0, stream>>>(
          X16 + (size_t)p * B * D, (int)D, Wp[p], (int)D, bp[p], (int)D,
          fused, nullptr, (int)(3 * BH), (int)(p * BH), nullptr, nullptr, 0);
    }
  }

  // Stage 2: fusion GEMM -> fcur [B, H] (f32), constant across time steps.
  {
    dim3 grid((unsigned)(B / 128), (unsigned)(H / 128));
    wmma_gemm_k<1><<<grid, 128, 0, stream>>>(
        fused, (int)(3 * BH), Wf16, (int)(3 * BH), bf, (int)(3 * BH), nullptr,
        fcur, (int)H, 0, nullptr, nullptr, 0);
  }

  // Stage 3: 16 LIF steps. Each: fusion-layer LIF (elementwise, spike f16),
  // then recurrent GEMM with fused integration-layer LIF epilogue.
  {
    const long nBH = (long)(B * H);
    const int lifBlocks = (int)((nBH + 255) / 256);
    dim3 grid((unsigned)(B / 128), (unsigned)(H / 128));
    for (int t = 0; t < NUM_STEPS; ++t) {
      lif_fusion_step_k<<<lifBlocks, 256, 0, stream>>>(fcur, fmem, fspk, nBH);
      wmma_gemm_k<2><<<grid, 128, 0, stream>>>(
          fspk, (int)H, Wi16, (int)H, bi, (int)H, nullptr, nullptr, 0, 0,
          imem, pooled, (int)H);
    }
  }

  // Stage 4: pooled mean (k/16 exact in f16) and output GEMM -> out f32.
  cvt(pooled, pooled16, 1.0f / (float)NUM_STEPS, (long)(B * H));
  {
    dim3 grid((unsigned)(B / 128), (unsigned)(O / 128));
    wmma_gemm_k<1><<<grid, 128, 0, stream>>>(
        pooled16, (int)H, Wo16, (int)H, bo, (int)H, nullptr, out, (int)O, 0,
        nullptr, nullptr, 0);
  }
}
